// LorentzFFN_77163382439946
// MI455X (gfx1250) — matrix operand order repaired and checked by reading, not compiled, not run on hardware
//
#include <hip/hip_runtime.h>
#include <math.h>

#define EPSV 1e-6f
#define BM 128
#define BN 128
#define BK 32
#define LDSS 40   // LDS row stride in bf16 units (80B: 16B-aligned, conflict-free)

typedef __attribute__((ext_vector_type(16))) __bf16 v16bf;
typedef __attribute__((ext_vector_type(8)))  float  v8f;
typedef __attribute__((ext_vector_type(4)))  float  f32x4;
typedef __attribute__((ext_vector_type(4)))  int    i32x4;

union FragBF { v16bf v; f32x4 f[2]; };

#if defined(__gfx1250__) && __has_builtin(__builtin_amdgcn_global_load_async_to_lds_b128)
#define ASYNC_COPY 1
#endif

__device__ __forceinline__ void async_cp16(const unsigned short* g, unsigned short* l) {
#if defined(ASYNC_COPY)
  // global -> LDS DMA, 16 bytes per lane, tracked by ASYNCcnt (no VGPR staging)
  __builtin_amdgcn_global_load_async_to_lds_b128(
      (__attribute__((address_space(1))) i32x4*)g,
      (__attribute__((address_space(3))) i32x4*)l, 0, 0);
#else
  (void)g; (void)l;
#endif
}

__device__ __forceinline__ void wait_async0() {
#if defined(ASYNC_COPY)
#if __has_builtin(__builtin_amdgcn_s_wait_asynccnt)
  __builtin_amdgcn_s_wait_asynccnt(0);
#else
  asm volatile("s_wait_asynccnt 0x0" ::: "memory");
#endif
#endif
}

__device__ __forceinline__ unsigned short f2bf(float x) {
  unsigned u  = __float_as_uint(x);
  unsigned rb = ((u >> 16) & 1u) + 0x7FFFu;   // round-to-nearest-even
  return (unsigned short)((u + rb) >> 16);
}

__device__ __forceinline__ float block_sum256(float v, float* sbuf) {
#pragma unroll
  for (int o = 16; o > 0; o >>= 1) v += __shfl_xor(v, o, 32);
  const int wave = threadIdx.x >> 5;
  if ((threadIdx.x & 31) == 0) sbuf[wave] = v;
  __syncthreads();
  float tot = 0.f;
#pragma unroll
  for (int i = 0; i < 8; ++i) tot += sbuf[i];
  __syncthreads();
  return tot;
}

// ---------------- weight transpose + f32->bf16 convert: in[R][C] -> out[C][R]
__global__ __launch_bounds__(256)
void transp_f32_bf16(const float* __restrict__ in, unsigned short* __restrict__ out,
                     int R, int C) {
  __shared__ float tile[32][33];
  const int bc = blockIdx.x * 32;
  const int br = blockIdx.y * 32;
  const int tx = threadIdx.x & 31;
  const int ty = threadIdx.x >> 5;  // 0..7
#pragma unroll
  for (int j = 0; j < 32; j += 8)
    tile[ty + j][tx] = in[(size_t)(br + ty + j) * C + bc + tx];
  __syncthreads();
#pragma unroll
  for (int j = 0; j < 32; j += 8)
    out[(size_t)(bc + ty + j) * R + br + tx] = f2bf(tile[tx][ty + j]);
}

// ---------------- log-map at mu=(1,0,...,0): x[row][1024] f32 -> v bf16
__global__ __launch_bounds__(256)
void logmap_kernel(const float* __restrict__ x, unsigned short* __restrict__ v, int D) {
  __shared__ float sbuf[8];
  const int row = blockIdx.x;
  const float* xr = x + (size_t)row * D;
  float loc[4];
  float ss = 0.f;
#pragma unroll
  for (int j = 0; j < 4; ++j) {
    const int i = threadIdx.x + 256 * j;
    float val = xr[i];
    if (i > 0) { val = fminf(fmaxf(val, -8.f), 8.f); ss += val * val; }
    loc[j] = val;
  }
  ss = block_sum256(ss, sbuf);
  const float y0    = sqrtf(1.f + ss + EPSV);
  const float my    = fmaxf(y0, 1.f + EPSV);          // = -xy
  const float d     = fmaxf(acoshf(my), 1e-3f);
  const float dirv0 = y0 - my;                        // y0 + xy
  const float dn    = sqrtf(fmaxf(ss - dirv0 * dirv0, EPSV));
  const float scale = d / (dn + EPSV);
  unsigned short* vr = v + (size_t)row * D;
#pragma unroll
  for (int j = 0; j < 4; ++j) {
    const int i = threadIdx.x + 256 * j;
    const float o = (i == 0) ? scale * dirv0 : scale * loc[j];
    vr[i] = f2bf(o);
  }
}

// ---------------- exp-map at mu + double project, in place on f32 u[row][1024]
__global__ __launch_bounds__(256)
void expmap_kernel(float* __restrict__ u, int D) {
  __shared__ float sbuf[8];
  const int row = blockIdx.x;
  float* ur = u + (size_t)row * D;
  float loc[4];
  float ss = 0.f;
#pragma unroll
  for (int j = 0; j < 4; ++j) {
    const int i = threadIdx.x + 256 * j;
    const float val = ur[i];
    loc[j] = val;
    if (i > 0) ss += val * val;      // v_tan0 == 0 at mu
  }
  ss = block_sum256(ss, sbuf);
  const float vn = fminf(sqrtf(ss + EPSV), 5.f);
  const float s  = sinhf(vn) / (vn + EPSV);
  float sc = 0.f;
#pragma unroll
  for (int j = 0; j < 4; ++j) {
    const int i = threadIdx.x + 256 * j;
    if (i > 0) {
      const float r = fminf(fmaxf(s * loc[j], -8.f), 8.f);
      loc[j] = r;
      sc += r * r;
    }
  }
  sc = block_sum256(sc, sbuf);
#pragma unroll
  for (int j = 0; j < 4; ++j) {
    const int i = threadIdx.x + 256 * j;
    ur[i] = (i == 0) ? sqrtf(1.f + sc + EPSV) : loc[j];
  }
}

// ---------------- bf16 WMMA GEMM: C[M][N] = A[M][K] * BT[N][K]^T (+bias, epi)
// Double-buffered LDS, one barrier per K-step; async global->LDS DMA on CDNA5.
// EPI==0: bias + exact GELU, write bf16. EPI==1: bias, write f32.
template <int EPI>
__global__ __launch_bounds__(256)
void gemm_bf16_wmma(const unsigned short* __restrict__ A,
                    const unsigned short* __restrict__ BT,
                    const float* __restrict__ bias,
                    void* __restrict__ Cout,
                    int M, int N, int K) {
  __shared__ unsigned short As[2][BM * LDSS];
  __shared__ unsigned short Bs[2][BN * LDSS];

  const int tid  = threadIdx.x;
  const int lane = tid & 31;
  const int wave = tid >> 5;
  const int wm   = wave >> 2;      // 0..1  (64-row half)
  const int wn   = wave & 3;       // 0..3  (32-col slice)
  const int l15  = lane & 15;
  const int half = lane >> 4;

  const int bm0 = blockIdx.x * BM;
  const int bn0 = blockIdx.y * BN;

  const int lr = tid >> 2;         // 0..63
  const int lc = (tid & 3) * 8;    // 0,8,16,24

  const unsigned short* gA0 = A  + (size_t)(bm0 + lr     ) * K + lc;
  const unsigned short* gA1 = A  + (size_t)(bm0 + lr + 64) * K + lc;
  const unsigned short* gB0 = BT + (size_t)(bn0 + lr     ) * K + lc;
  const unsigned short* gB1 = BT + (size_t)(bn0 + lr + 64) * K + lc;
  const int sOff0 = (lr     ) * LDSS + lc;
  const int sOff1 = (lr + 64) * LDSS + lc;

  v8f acc[4][2] = {};
  const int nt = K / BK;

#if defined(ASYNC_COPY)
  // prologue: DMA tile 0 into buffer 0
  async_cp16(gA0, &As[0][sOff0]);
  async_cp16(gA1, &As[0][sOff1]);
  async_cp16(gB0, &Bs[0][sOff0]);
  async_cp16(gB1, &Bs[0][sOff1]);
#else
  {
    const f32x4 ra0 = *(const f32x4*)gA0;
    const f32x4 ra1 = *(const f32x4*)gA1;
    const f32x4 rb0 = *(const f32x4*)gB0;
    const f32x4 rb1 = *(const f32x4*)gB1;
    *(f32x4*)&As[0][sOff0] = ra0;
    *(f32x4*)&As[0][sOff1] = ra1;
    *(f32x4*)&Bs[0][sOff0] = rb0;
    *(f32x4*)&Bs[0][sOff1] = rb1;
  }
#endif

  for (int it = 0; it < nt; ++it) {
    const int cur = it & 1;
    const int nxt = cur ^ 1;

#if defined(ASYNC_COPY)
    wait_async0();           // my DMAs into buf[cur] have landed in LDS
    __syncthreads();         // everyone's landed; everyone done reading buf[nxt]
    if (it + 1 < nt) {
      const int k1 = (it + 1) * BK;
      async_cp16(gA0 + k1, &As[nxt][sOff0]);
      async_cp16(gA1 + k1, &As[nxt][sOff1]);
      async_cp16(gB0 + k1, &Bs[nxt][sOff0]);
      async_cp16(gB1 + k1, &Bs[nxt][sOff1]);
    }
#else
    __syncthreads();         // buf[cur] stores visible; buf[nxt] free to refill
    f32x4 ra0, ra1, rb0, rb1;
    if (it + 1 < nt) {
      const int k1 = (it + 1) * BK;
      ra0 = *(const f32x4*)(gA0 + k1);
      ra1 = *(const f32x4*)(gA1 + k1);
      rb0 = *(const f32x4*)(gB0 + k1);
      rb1 = *(const f32x4*)(gB1 + k1);
    }
#endif

    const unsigned short* AsC = &As[cur][0];
    const unsigned short* BsC = &Bs[cur][0];

    FragBF af[4], bf[2];
#pragma unroll
    for (int mi = 0; mi < 4; ++mi) {
      // A 16x32 bf16: lanes 0-15 hold K 0..7 / 16..23 of row l15; lanes 16-31 +8
      const unsigned short* p = AsC + (wm * 64 + mi * 16 + l15) * LDSS + 8 * half;
      af[mi].f[0] = *(const f32x4*)(p);
      af[mi].f[1] = *(const f32x4*)(p + 16);
    }
#pragma unroll
    for (int ni = 0; ni < 2; ++ni) {
      // B 32x16 bf16: lanes 0-15 hold K 0..7 / 16..23 of col l15; lanes 16-31 +8
      const unsigned short* p = BsC + (wn * 32 + ni * 16 + l15) * LDSS + 8 * half;
      bf[ni].f[0] = *(const f32x4*)(p);
      bf[ni].f[1] = *(const f32x4*)(p + 16);
    }
#pragma unroll
    for (int mi = 0; mi < 4; ++mi)
#pragma unroll
      for (int ni = 0; ni < 2; ++ni)
        acc[mi][ni] = __builtin_amdgcn_wmma_f32_16x16x32_bf16(
            false, af[mi].v, false, bf[ni].v, (short)0, acc[mi][ni], false, false);

#if !defined(ASYNC_COPY)
    if (it + 1 < nt) {
      *(f32x4*)&As[nxt][sOff0] = ra0;
      *(f32x4*)&As[nxt][sOff1] = ra1;
      *(f32x4*)&Bs[nxt][sOff0] = rb0;
      *(f32x4*)&Bs[nxt][sOff1] = rb1;
    }
#endif
  }

  // C/D layout: lane = N within tile; VGPR r -> M = (lane>=16 ? 8 : 0)+r
  const int mrow0 = half * 8;
#pragma unroll
  for (int ni = 0; ni < 2; ++ni) {
    const int gn = bn0 + wn * 32 + ni * 16 + l15;
    const float bv = bias[gn];
#pragma unroll
    for (int mi = 0; mi < 4; ++mi) {
      const int gm0 = bm0 + wm * 64 + mi * 16 + mrow0;
#pragma unroll
      for (int r = 0; r < 8; ++r) {
        float c = acc[mi][ni][r] + bv;
        if (EPI == 0) {
          c = 0.5f * c * (1.0f + erff(c * 0.70710678118654752f));  // exact GELU
          ((unsigned short*)Cout)[(size_t)(gm0 + r) * N + gn] = f2bf(c);
        } else {
          ((float*)Cout)[(size_t)(gm0 + r) * N + gn] = c;
        }
      }
    }
  }
}

extern "C" void kernel_launch(void* const* d_in, const int* in_sizes, int n_in,
                              void* d_out, int out_size, void* d_ws, size_t ws_size,
                              hipStream_t stream) {
  const float* x  = (const float*)d_in[0];
  const float* W1 = (const float*)d_in[1];
  const float* b1 = (const float*)d_in[2];
  const float* W2 = (const float*)d_in[3];
  const float* b2 = (const float*)d_in[4];

  const int H = in_sizes[2];        // 2048
  const int D = in_sizes[1] / H;    // 1024
  const int M = in_sizes[0] / D;    // 16384 (= B*T)

  unsigned char* ws = (unsigned char*)d_ws;
  unsigned short* vbf = (unsigned short*)ws;                                  // M*D bf16
  unsigned short* hbf = (unsigned short*)(ws + (size_t)M * D * 2);            // M*H bf16
  unsigned short* w1t = (unsigned short*)(ws + (size_t)M * D * 2 + (size_t)M * H * 2);
  unsigned short* w2t = w1t + (size_t)H * D;
  float* out = (float*)d_out;

  // weights: W1[D][H] -> W1T[H][D] bf16 ; W2[H][D] -> W2T[D][H] bf16
  transp_f32_bf16<<<dim3(H / 32, D / 32), 256, 0, stream>>>(W1, w1t, D, H);
  transp_f32_bf16<<<dim3(D / 32, H / 32), 256, 0, stream>>>(W2, w2t, H, D);

  // v = log_map(mu, x)  (bf16)
  logmap_kernel<<<M, 256, 0, stream>>>(x, vbf, D);

  // h = gelu(v @ W1 + b1)  (bf16)
  gemm_bf16_wmma<0><<<dim3(M / BM, H / BN), 256, 0, stream>>>(vbf, w1t, b1, (void*)hbf, M, H, D);

  // u = h @ W2 + b2  (f32, directly into d_out)
  gemm_bf16_wmma<1><<<dim3(M / BM, D / BN), 256, 0, stream>>>(hbf, w2t, b2, (void*)out, M, D, H);

  // out = project(exp_map(mu, u))  in place
  expmap_kernel<<<M, 256, 0, stream>>>(out, D);
}
